// Decoder_72189810311180
// MI455X (gfx1250) — matrix operand order repaired and checked by reading, not compiled, not run on hardware
//
#include <hip/hip_runtime.h>
#include <hip/hip_bf16.h>
#include <stdint.h>

// Problem constants
#define BSZ 16
#define TSEQ 128
#define HDIM 512
#define VOC 32000
#define NSTEP (TSEQ - 1)   // 127 decode steps

typedef __attribute__((ext_vector_type(16))) __bf16 v16bf;
typedef __attribute__((ext_vector_type(8)))  __bf16 v8bf;
typedef __attribute__((ext_vector_type(8)))  float  v8f;
typedef __attribute__((ext_vector_type(4)))  unsigned int v4u;
typedef __attribute__((ext_vector_type(8)))  int v8i;
typedef __attribute__((ext_vector_type(4)))  int v4i;

// TDM is a device-side amdgcn builtin; host pass parses __global__ bodies too,
// so gate on the device compilation pass (builtin confirmed present by probes).
#if defined(__HIP_DEVICE_COMPILE__)
#  define HAVE_TDM 1
#endif

// ---------------- helpers ----------------

// Build 16x32 bf16 WMMA A-operand fragment from a row of X (bf16, row-major):
// lanes 0-15 (m=lane): K = {kb..kb+7, kb+16..kb+23} with kb = 0
// lanes 16-31 (m=lane-16): kb = 8
__device__ __forceinline__ v16bf ldA16(const __bf16* rowChunk, int kbase) {
  v8bf lo = *(const v8bf*)(rowChunk + kbase);
  v8bf hi = *(const v8bf*)(rowChunk + kbase + 16);
  return __builtin_shufflevector(lo, hi, 0,1,2,3,4,5,6,7,8,9,10,11,12,13,14,15);
}

// Same but from f32 memory (on-the-fly bf16 convert)
__device__ __forceinline__ v16bf ldA32(const float* rowChunk, int kbase) {
  v16bf a;
#pragma unroll
  for (int i = 0; i < 8; ++i) {
    a[i]     = (__bf16)rowChunk[kbase + i];
    a[i + 8] = (__bf16)rowChunk[kbase + 16 + i];
  }
  return a;
}

__device__ __forceinline__ v8f wmma_bf16(v16bf a, v16bf b, v8f c) {
  // D = A(16x32) * B(32x16) + C(16x16), f32 accumulate
  return __builtin_amdgcn_wmma_f32_16x16x32_bf16(false, a, false, b, (short)0, c, false, false);
}

__device__ __forceinline__ float sigmoidf_(float x) {
  return 1.0f / (1.0f + __expf(-x));
}

// ---------------- pre-pack kernels ----------------

// Pack W[N,K] f32 row-major into WMMA B-operand tile layout (bf16):
// Wp[((nt*KC + kc)*32 + lane)*16 + i] = W[nt*16 + lane%16, kc*32 + 8*(lane/16) + (i&7) + 16*(i>>3)]
__global__ void pack_b_kernel(const float* __restrict__ W, __bf16* __restrict__ Wp,
                              int N, int K) {
  int KC = K >> 5;
  long long total = (long long)N * K;
  long long stride = (long long)gridDim.x * blockDim.x;
  for (long long e = (long long)blockIdx.x * blockDim.x + threadIdx.x; e < total; e += stride) {
    int i    = (int)(e & 15);
    int lane = (int)((e >> 4) & 31);
    long long cnk = e >> 9;
    int kc = (int)(cnk % KC);
    int nt = (int)(cnk / KC);
    Wp[e] = (__bf16)W[(long long)(nt * 16 + (lane & 15)) * K + (kc * 32 + ((lane >> 4) << 3) + (i & 7) + ((i >> 3) << 4))];
  }
}

__global__ void f32_to_bf16_kernel(const float* __restrict__ in, __bf16* __restrict__ out,
                                   long long n) {
  long long stride = (long long)gridDim.x * blockDim.x;
  for (long long e = (long long)blockIdx.x * blockDim.x + threadIdx.x; e < n; e += stride)
    out[e] = (__bf16)in[e];
}

// dec_outputs[:, 0, :] must be zero (harness poisons d_out)
__global__ void zero_t0_kernel(float* __restrict__ out) {
  long long total = (long long)BSZ * VOC;
  long long stride = (long long)gridDim.x * blockDim.x;
  for (long long e = (long long)blockIdx.x * blockDim.x + threadIdx.x; e < total; e += stride) {
    long long b = e / VOC, v = e % VOC;
    out[b * (long long)TSEQ * VOC + v] = 0.0f;
  }
}

// ---------------- vt = enc_out @ Wv^T + bv  (hoisted, loop-invariant) ----------------

__global__ void __launch_bounds__(256) vt_kernel(const __bf16* __restrict__ encbf,
                                                 const __bf16* __restrict__ Wvp,
                                                 const float* __restrict__ bv,
                                                 float* __restrict__ vt) {
  int mt   = blockIdx.x;           // 128 M-tiles over B*T = 2048 rows
  int lane = threadIdx.x & 31;
  int wave = threadIdx.x >> 5;     // 8 waves
  int m = lane & 15, kbase = (lane >> 4) * 8;
  const __bf16* arow = encbf + (long long)(mt * 16 + m) * HDIM;

  for (int j = 0; j < 2; ++j) {    // 32 N-tiles / 8 waves, 2 at a time
    int nt0 = wave * 4 + 2 * j;
    const __bf16* bp0 = Wvp + nt0 * 8192;
    v8f acc0 = {}, acc1 = {};
#pragma unroll
    for (int kc = 0; kc < 16; ++kc) {
      v16bf a  = ldA16(arow + kc * 32, kbase);
      v16bf b0 = *(const v16bf*)(bp0 + kc * 512 + lane * 16);
      v16bf b1 = *(const v16bf*)(bp0 + 8192 + kc * 512 + lane * 16);
      acc0 = wmma_bf16(a, b0, acc0);
      acc1 = wmma_bf16(a, b1, acc1);
    }
    int col0 = nt0 * 16 + (lane & 15);
    float bias0 = bv[col0], bias1 = bv[col0 + 16];
#pragma unroll
    for (int r = 0; r < 8; ++r) {
      int row = mt * 16 + r + ((lane >> 4) << 3);
      vt[(long long)row * HDIM + col0]      = acc0[r] + bias0;
      vt[(long long)row * HDIM + col0 + 16] = acc1[r] + bias1;
    }
  }
}

// ---------------- sequential recurrence: attention + GRU, one persistent WG ----------------

__global__ void __launch_bounds__(512) recurrence_kernel(
    const float* __restrict__ enc, const float* __restrict__ enc_hid,
    const int* __restrict__ mask, const int* __restrict__ target,
    const float* __restrict__ emb,
    const __bf16* __restrict__ Wqp, const float* __restrict__ bq,
    const float* __restrict__ Wc, const float* __restrict__ bc,
    const __bf16* __restrict__ Wihp, const __bf16* __restrict__ Whhp,
    const float* __restrict__ bih, const float* __restrict__ bhh,
    const float* __restrict__ vt, __bf16* __restrict__ hpack,
    float* __restrict__ outHT) {
  __shared__ __bf16 sH[16 * 512];    // current hidden (bf16)      16 KB
  __shared__ __bf16 sCtx[16 * 512];  // attention context          16 KB
  __shared__ __bf16 sQ[16 * 512];    // q, reused as h_new stage   16 KB
  __shared__ float  sAw[16 * 128];   // scores -> softmax weights   8 KB
  __shared__ int    sTok[16];

  int tid = threadIdx.x, lane = tid & 31, wave = tid >> 5;  // 16 waves
  int m = lane & 15, kbase = (lane >> 4) * 8;

  for (int e = tid; e < 16 * 512; e += 512) sH[e] = (__bf16)enc_hid[e];
  __syncthreads();

  for (int s = 0; s < NSTEP; ++s) {
    // tokens for this step: BOS at s==0 else target[:, s]
    if (tid < 16) sTok[tid] = (s == 0) ? 1 : target[tid * TSEQ + s];

    // ---- q = h @ Wq^T + bq, WMMA (M=16 batch tile) ----
    {
      int nt0 = wave;                // tiles {wave, wave+16}
      v8f acc0 = {}, acc1 = {};
#pragma unroll
      for (int kc = 0; kc < 16; ++kc) {
        v16bf a  = ldA16(&sH[m * 512 + kc * 32], kbase);
        v16bf b0 = *(const v16bf*)(Wqp + (nt0 * 16 + kc) * 512 + lane * 16);
        v16bf b1 = *(const v16bf*)(Wqp + ((nt0 + 16) * 16 + kc) * 512 + lane * 16);
        acc0 = wmma_bf16(a, b0, acc0);
        acc1 = wmma_bf16(a, b1, acc1);
      }
      int col0 = nt0 * 16 + (lane & 15);
      float bias0 = bq[col0], bias1 = bq[col0 + 256];
#pragma unroll
      for (int r = 0; r < 8; ++r) {
        int row = r + ((lane >> 4) << 3);
        sQ[row * 512 + col0]       = (__bf16)(acc0[r] + bias0);
        sQ[row * 512 + col0 + 256] = (__bf16)(acc1[r] + bias1);
      }
    }
    __syncthreads();

    // ---- scores[b,t] = tanh(q[b]+vt[b,t]) . Wc + bc  (masked) ----
    {
      float bc0 = bc[0];
      for (int task = tid; task < BSZ * TSEQ; task += 512) {
        int b = task >> 7, t = task & 127;
        float sc;
        if (mask[b * TSEQ + t] == 0) {
          sc = -3.4028234663852886e38f;
        } else {
          sc = bc0;
          const float*  vrow = vt + (long long)(b * TSEQ + t) * HDIM;
          const __bf16* qrow = &sQ[b * 512];
          for (int k = 0; k < HDIM; ++k)
            sc += tanhf((float)qrow[k] + vrow[k]) * Wc[k];
        }
        sAw[task] = sc;
      }
    }
    __syncthreads();

    // ---- softmax over t: wave b handles batch b (wave32 shuffles) ----
    {
      int b = wave;
      float v0 = sAw[b * 128 + lane],      v1 = sAw[b * 128 + 32 + lane];
      float v2 = sAw[b * 128 + 64 + lane], v3 = sAw[b * 128 + 96 + lane];
      float mx = fmaxf(fmaxf(v0, v1), fmaxf(v2, v3));
      for (int off = 16; off; off >>= 1) mx = fmaxf(mx, __shfl_xor(mx, off, 32));
      float e0 = __expf(v0 - mx), e1 = __expf(v1 - mx);
      float e2 = __expf(v2 - mx), e3 = __expf(v3 - mx);
      float sm = e0 + e1 + e2 + e3;
      for (int off = 16; off; off >>= 1) sm += __shfl_xor(sm, off, 32);
      float inv = 1.0f / sm;
      sAw[b * 128 + lane]      = e0 * inv; sAw[b * 128 + 32 + lane] = e1 * inv;
      sAw[b * 128 + 64 + lane] = e2 * inv; sAw[b * 128 + 96 + lane] = e3 * inv;
    }
    __syncthreads();

    // ---- ctx[b,j] = sum_t aw[b,t] * enc[b,t,j] ----
    for (int idx = tid; idx < 16 * 512; idx += 512) {
      int b = idx >> 9, j = idx & 511;
      float acc = 0.0f;
      const float* erow = enc + (long long)(b * TSEQ) * HDIM + j;
      const float* arow = &sAw[b * 128];
      for (int t = 0; t < TSEQ; ++t) acc += arow[t] * erow[t * HDIM];
      sCtx[idx] = (__bf16)acc;
    }
    __syncthreads();

    // ---- GRU: gi = [xe,ctx] @ Wih^T, gh = h @ Whh^T ; fused per output tile ----
    for (int tl = 0; tl < 2; ++tl) {
      int nt = wave + tl * 16;       // 32 output tiles over H
      v8f aR = {}, aZ = {}, aIN = {}, aHN = {};
      const float* embrow = emb + (long long)sTok[m] * HDIM;  // xe row per lane's m
      // input-side: K = 1024 (xe: kc 0..15 from emb f32, ctx: kc 16..31 from LDS)
      for (int kc = 0; kc < 32; ++kc) {
        v16bf a = (kc < 16) ? ldA32(embrow + kc * 32, kbase)
                            : ldA16(&sCtx[m * 512 + (kc - 16) * 32], kbase);
        v16bf bR = *(const v16bf*)(Wihp + ((nt * 32 + kc) * 512 + lane * 16));
        v16bf bZ = *(const v16bf*)(Wihp + (((nt + 32) * 32 + kc) * 512 + lane * 16));
        v16bf bN = *(const v16bf*)(Wihp + (((nt + 64) * 32 + kc) * 512 + lane * 16));
        aR  = wmma_bf16(a, bR, aR);
        aZ  = wmma_bf16(a, bZ, aZ);
        aIN = wmma_bf16(a, bN, aIN);
      }
      // hidden-side: K = 512 (ir+hr / iz+hz share accumulators; hn separate)
#pragma unroll
      for (int kc = 0; kc < 16; ++kc) {
        v16bf a = ldA16(&sH[m * 512 + kc * 32], kbase);
        v16bf bR = *(const v16bf*)(Whhp + ((nt * 16 + kc) * 512 + lane * 16));
        v16bf bZ = *(const v16bf*)(Whhp + (((nt + 32) * 16 + kc) * 512 + lane * 16));
        v16bf bN = *(const v16bf*)(Whhp + (((nt + 64) * 16 + kc) * 512 + lane * 16));
        aR  = wmma_bf16(a, bR, aR);
        aZ  = wmma_bf16(a, bZ, aZ);
        aHN = wmma_bf16(a, bN, aHN);
      }
      int col = nt * 16 + (lane & 15);
      float biR  = bih[col] + bhh[col];
      float biZ  = bih[512 + col] + bhh[512 + col];
      float biIN = bih[1024 + col];
      float biHN = bhh[1024 + col];
#pragma unroll
      for (int r = 0; r < 8; ++r) {
        int row = r + ((lane >> 4) << 3);
        float rr = sigmoidf_(aR[r] + biR);
        float zz = sigmoidf_(aZ[r] + biZ);
        float nn = tanhf(aIN[r] + biIN + rr * (aHN[r] + biHN));
        float ho = (float)sH[row * 512 + col];
        sQ[row * 512 + col] = (__bf16)((1.0f - zz) * nn + zz * ho);  // stage h_new in sQ
      }
    }
    __syncthreads();

    // ---- commit: sQ -> sH, and emit h_t to global in WMMA A-operand layout ----
    {
      __bf16* hp = hpack + (long long)s * 8192;
      for (int e = tid; e < 16 * 512; e += 512) {
        __bf16 hv = sQ[e];
        sH[e] = hv;
        int mm = e >> 9, k = e & 511;
        int kc = k >> 5, kk = k & 31;
        int l2 = mm + (((kk >> 3) & 1) << 4);
        int ii = (kk & 7) + ((kk >= 16) ? 8 : 0);
        hp[(kc * 32 + l2) * 16 + ii] = hv;
        if (s == NSTEP - 1) outHT[e] = (float)hv;
      }
    }
    __syncthreads();
  }
}

// ---------------- deferred output projection: [127*16, 512] @ [512, 32000] ----------------

__global__ void __launch_bounds__(256) logits_kernel(const __bf16* __restrict__ hpack,
                                                     const __bf16* __restrict__ Wop,
                                                     const float* __restrict__ bo,
                                                     float* __restrict__ out) {
  __shared__ __bf16 sA[8192];        // one step's A-panel (16x512 bf16) = 16 KB
  int s      = blockIdx.x;           // decode step 0..126 -> output t = s+1
  int ntBase = blockIdx.y * 80;      // 2000 N-tiles / 25 blocks
  int tid = threadIdx.x, lane = tid & 31, wave = tid >> 5;  // 8 waves
  const __bf16* src = hpack + (long long)s * 8192;

#if HAVE_TDM
  if (wave == 0) {
    // 1-D TDM descriptor: contiguous 16 KB tile, data_size = 8B -> 2048 units
    unsigned lds_off = (unsigned)(unsigned long long)(uintptr_t)&sA[0];
    unsigned long long ga = (unsigned long long)(uintptr_t)src;
    v4u g0 = { 1u,                                  // count=1, user descriptor
               lds_off,                              // lds_addr[31:0]
               (unsigned)ga,                         // global_addr[31:0]
               ((unsigned)((ga >> 32) & 0x1FFFFFFull)) | (2u << 30) };  // addr[56:32], type=2
    v8i g1 = { (int)(3u << 16),        // data_size=8B; workgroup_mask=0
               (int)(2048u << 16),     // tensor_dim0[15:0] in bits 63:48
               (int)(1u << 16),        // tensor_dim0 hi=0, tensor_dim1=1
               (int)(2048u << 16),     // tile_dim0 = 2048 units (16 KB)
               0,                      // tile_dim1/2 unused
               2048,                   // tensor_dim0_stride
               0, 0 };
    v4i z4 = { 0, 0, 0, 0 };
#if __clang_major__ >= 23
    v8i z8 = { 0, 0, 0, 0, 0, 0, 0, 0 };
    __builtin_amdgcn_tensor_load_to_lds(g0, g1, z4, z4, z8, 0);
#else
    __builtin_amdgcn_tensor_load_to_lds(g0, g1, z4, z4, 0);
#endif
    __builtin_amdgcn_s_wait_tensorcnt(0);
  }
  __syncthreads();
#else
  {
    int base = tid * 32;
    for (int j = 0; j < 32; ++j) sA[base + j] = src[base + j];
  }
  __syncthreads();
#endif

  // 80 N-tiles / 8 waves; 2 tiles at a time -> two independent WMMA chains
  for (int t = 0; t < 5; ++t) {
    int nt0 = ntBase + wave * 10 + 2 * t;
    const __bf16* bp0 = Wop + (long long)nt0 * 8192;
    v8f acc0 = {}, acc1 = {};
#pragma unroll
    for (int kc = 0; kc < 16; ++kc) {
      v16bf a  = *(const v16bf*)(sA + (kc * 32 + lane) * 16);   // shared A fragment
      v16bf b0 = *(const v16bf*)(bp0 + kc * 512 + lane * 16);
      v16bf b1 = *(const v16bf*)(bp0 + 8192 + kc * 512 + lane * 16);
      acc0 = wmma_bf16(a, b0, acc0);
      acc1 = wmma_bf16(a, b1, acc1);
    }
    if (t < 4) {
      __builtin_prefetch(bp0 + 16384, 0, 0);   // next pair of B panels
      __builtin_prefetch(bp0 + 24576, 0, 0);
    }
    int n0 = nt0 * 16 + (lane & 15);
    float bias0 = bo[n0], bias1 = bo[n0 + 16];
#pragma unroll
    for (int r = 0; r < 8; ++r) {
      int bb = r + ((lane >> 4) << 3);             // batch row
      long long rowOff = (long long)(bb * TSEQ + (s + 1)) * VOC;
      __builtin_nontemporal_store(acc0[r] + bias0, &out[rowOff + n0]);
      __builtin_nontemporal_store(acc1[r] + bias1, &out[rowOff + n0 + 16]);
    }
  }
}

// ---------------- host launcher ----------------

extern "C" void kernel_launch(void* const* d_in, const int* in_sizes, int n_in,
                              void* d_out, int out_size, void* d_ws, size_t ws_size,
                              hipStream_t stream) {
  (void)in_sizes; (void)n_in; (void)out_size; (void)ws_size;
  const float* enc  = (const float*)d_in[0];
  const float* ehid = (const float*)d_in[1];
  const int*   mask = (const int*)d_in[2];
  const int*   targ = (const int*)d_in[3];
  const float* emb  = (const float*)d_in[4];
  const float* Wq   = (const float*)d_in[5];
  const float* bq   = (const float*)d_in[6];
  const float* Wv   = (const float*)d_in[7];
  const float* bv   = (const float*)d_in[8];
  const float* Wc   = (const float*)d_in[9];
  const float* bc   = (const float*)d_in[10];
  const float* Wih  = (const float*)d_in[11];
  const float* Whh  = (const float*)d_in[12];
  const float* bih  = (const float*)d_in[13];
  const float* bhh  = (const float*)d_in[14];
  const float* Wo   = (const float*)d_in[15];
  const float* bo   = (const float*)d_in[16];
  float* out = (float*)d_out;

  char* ws = (char*)d_ws;
  __bf16* Wop  = (__bf16*)(ws + 0);          // 32,768,000 B
  __bf16* Wihp = (__bf16*)(ws + 32768000);   //  3,145,728 B
  __bf16* Whhp = (__bf16*)(ws + 35913728);   //  1,572,864 B
  __bf16* Wqp  = (__bf16*)(ws + 37486592);   //    524,288 B
  __bf16* Wvp  = (__bf16*)(ws + 38010880);   //    524,288 B
  __bf16* encb = (__bf16*)(ws + 38535168);   //  2,097,152 B
  float*  vt   = (float*) (ws + 40632320);   //  4,194,304 B
  __bf16* hpk  = (__bf16*)(ws + 44826624);   //  2,080,768 B  (127 x 16x512 A-packed)

  pack_b_kernel<<<4096, 256, 0, stream>>>(Wo,  Wop,  VOC,      HDIM);
  pack_b_kernel<<<512,  256, 0, stream>>>(Wih, Wihp, 3 * HDIM, 2 * HDIM);
  pack_b_kernel<<<256,  256, 0, stream>>>(Whh, Whhp, 3 * HDIM, HDIM);
  pack_b_kernel<<<128,  256, 0, stream>>>(Wq,  Wqp,  HDIM,     HDIM);
  pack_b_kernel<<<128,  256, 0, stream>>>(Wv,  Wvp,  HDIM,     HDIM);
  f32_to_bf16_kernel<<<1024, 256, 0, stream>>>(enc, encb, (long long)BSZ * TSEQ * HDIM);

  vt_kernel<<<128, 256, 0, stream>>>(encb, Wvp, bv, vt);
  zero_t0_kernel<<<512, 256, 0, stream>>>(out);

  recurrence_kernel<<<1, 512, 0, stream>>>(enc, ehid, mask, targ, emb, Wqp, bq, Wc, bc,
                                           Wihp, Whhp, bih, bhh, vt, hpk,
                                           out + (long long)BSZ * TSEQ * VOC);

  logits_kernel<<<dim3(127, 25), 256, 0, stream>>>(hpk, Wop, bo, out);
}
